// GTrXLMultiHeadSelfAttention_17686675325075
// MI455X (gfx1250) — matrix operand order repaired and checked by reading, not compiled, not run on hardware
//
#include <hip/hip_runtime.h>

// GTrXL multi-head self-attention for MI455X (gfx1250, wave32, WMMA + TDM).
// S=2048 B=2 D=1024 H=16 dk=64.  bf16 WMMA (16x16x32) with f32 accumulation.
// Dense GEMMs stage weight panels into LDS with the Tensor Data Mover
// (tensor_load_to_lds, TENSORcnt double-buffering); flash attention keeps
// direct global fragment loads (already contiguous) + global prefetch.

#define S_LEN  2048
#define BATCH  2
#define DMODEL 1024
#define NHEAD  16
#define DKH    64
#define ROWS   (S_LEN * BATCH)   // 4096 (row = s*B + b)
#define E3     (3 * DMODEL)      // 3072
#define GEMM_BN 256              // cols per block in dense GEMMs
#define GEMM_KT 32               // K step (WMMA K)

typedef __attribute__((ext_vector_type(16))) __bf16       v16bf;
typedef __attribute__((ext_vector_type(8)))  float        v8f;
typedef __attribute__((ext_vector_type(4)))  unsigned int u32x4;
typedef __attribute__((ext_vector_type(8)))  int          i32x8;
typedef __attribute__((ext_vector_type(4)))  int          i32x4;

union FragBits { u32x4 u[2]; v16bf v; };

__device__ __forceinline__ unsigned short f2bf(float f) {
  unsigned int u = __float_as_uint(f);
  u += 0x7FFFu + ((u >> 16) & 1u);          // round-to-nearest-even
  return (unsigned short)(u >> 16);
}

__device__ __forceinline__ v16bf load_frag_pair(const unsigned short* p0,
                                                const unsigned short* p1) {
  FragBits f;
  f.u[0] = *(const u32x4*)p0;
  f.u[1] = *(const u32x4*)p1;
  return f.v;
}

// A fragment: 16(M) x 32(K), source row-major, leading dim `ld` (elems).
// Lane<16 -> M=lane, K in {0..7,16..23}; lane>=16 -> M=lane-16, K in {8..15,24..31}.
__device__ __forceinline__ v16bf load_a_frag(const unsigned short* base, int ld, int lane) {
  const int half = lane >> 4, m = lane & 15;
  const unsigned short* r = base + (size_t)m * ld + half * 8;
  return load_frag_pair(r, r + 16);
}

// B fragment: 32(K) x 16(N) with B[k][n] = W[n][k]; W row-major `ld` (elems).
// Lane<16 -> N=lane, K in {0..7,16..23}; lane>=16 -> N=lane-16, K in {8..15,24..31}.
__device__ __forceinline__ v16bf load_b_frag(const unsigned short* base, int ld, int lane) {
  const int half = lane >> 4, n = lane & 15;
  const unsigned short* r = base + (size_t)n * ld + half * 8;
  return load_frag_pair(r, r + 16);
}

__device__ __forceinline__ v8f wmma_bf16(v16bf a, v16bf b, v8f c) {
  return __builtin_amdgcn_wmma_f32_16x16x32_bf16(false, a, false, b, (short)0, c,
                                                 false, false);
}

// ----------------------------------------------------------------- TDM
#if __has_builtin(__builtin_amdgcn_tensor_load_to_lds)
#define USE_TDM 1
#else
#define USE_TDM 0
#endif

#if USE_TDM
// Issue a TDM load of a [GEMM_BN rows x 32 elems] bf16 tile (rows `ld`=Kdim
// apart in global) into LDS at byte offset lds_off (rows packed [256][32]).
// D# layout per CDNA5 ISA 8.3/8.4. Uniform args -> SGPR descriptor groups.
__device__ __forceinline__ void tdm_load_panel(const unsigned short* gptr,
                                               unsigned lds_off,
                                               int kdim, int ncols) {
  const unsigned long long ga = (unsigned long long)gptr;
  u32x4 g0;
  g0[0] = 1u;                                        // count=1, user mode
  g0[1] = lds_off;                                   // LDS byte address
  g0[2] = (unsigned)(ga & 0xFFFFFFFFu);              // global_addr[31:0]
  g0[3] = (unsigned)((ga >> 32) & 0x01FFFFFFu) | (2u << 30); // ga[56:32] | type=2
  i32x8 g1;
  g1[0] = (int)(1u << 16);                           // data_size=1 (2 bytes)
  g1[1] = (int)(((unsigned)kdim & 0xFFFFu) << 16);   // tensor_dim0[15:0]
  g1[2] = (int)((((unsigned)kdim >> 16) & 0xFFFFu) |
                (((unsigned)ncols & 0xFFFFu) << 16)); // td0[31:16] | td1[15:0]
  g1[3] = (int)((((unsigned)ncols >> 16) & 0xFFFFu) |
                ((unsigned)GEMM_KT << 16));          // td1[31:16] | tile_dim0=32
  g1[4] = (int)GEMM_BN;                              // tile_dim1=256, tile_dim2=0
  g1[5] = kdim;                                      // tensor_dim0_stride[31:0]
  g1[6] = 0;                                         // stride0[47:32] | stride1[15:0]
  g1[7] = 0;
  i32x4 z4 = {0, 0, 0, 0};
#if __has_include(<hip/amd_detail/amd_gfx1250_TDM.h>)
  i32x8 z8 = {0, 0, 0, 0, 0, 0, 0, 0};
  __builtin_amdgcn_tensor_load_to_lds(g0, g1, z4, z4, z8, 0);  // 6-arg toolchain
#else
  __builtin_amdgcn_tensor_load_to_lds(g0, g1, z4, z4, 0);      // 5-arg toolchain
#endif
}
#endif

// Shared GEMM mainloop: acc[4] = A[mBase:16, :] * W[nBlock + wave*64 : +64, :]^T
// ldsB: [2][GEMM_BN][32] bf16 double buffer.
__device__ __forceinline__ void gemm_mainloop(const unsigned short* __restrict__ A,
                                              const unsigned short* __restrict__ W,
                                              int Kdim, int Ncols,
                                              int mBase, int nBlock,
                                              unsigned short (*ldsB)[GEMM_BN][GEMM_KT],
                                              int lane, int wave, v8f acc[4]) {
#if USE_TDM
  const unsigned lds0 = (unsigned)(unsigned long long)(void*)&ldsB[0][0][0];
  const unsigned lds1 = (unsigned)(unsigned long long)(void*)&ldsB[1][0][0];
  if (wave == 0)
    tdm_load_panel(W + (size_t)nBlock * Kdim, lds0, Kdim, Ncols);
  int buf = 0;
  for (int k = 0; k < Kdim; k += GEMM_KT, buf ^= 1) {
    if (wave == 0) {
      if (k + GEMM_KT < Kdim) {
        tdm_load_panel(W + (size_t)nBlock * Kdim + (k + GEMM_KT),
                       buf ? lds0 : lds1, Kdim, Ncols);
        __builtin_amdgcn_s_wait_tensorcnt(1);  // in-order: current panel done
      } else {
        __builtin_amdgcn_s_wait_tensorcnt(0);
      }
    }
    __syncthreads();                           // panel visible to all waves
    v16bf a  = load_a_frag(A + (size_t)mBase * Kdim + k, Kdim, lane);
    v16bf b0 = load_b_frag(&ldsB[buf][wave * 64 +  0][0], GEMM_KT, lane);
    v16bf b1 = load_b_frag(&ldsB[buf][wave * 64 + 16][0], GEMM_KT, lane);
    v16bf b2 = load_b_frag(&ldsB[buf][wave * 64 + 32][0], GEMM_KT, lane);
    v16bf b3 = load_b_frag(&ldsB[buf][wave * 64 + 48][0], GEMM_KT, lane);
    acc[0] = wmma_bf16(a, b0, acc[0]);
    acc[1] = wmma_bf16(a, b1, acc[1]);
    acc[2] = wmma_bf16(a, b2, acc[2]);
    acc[3] = wmma_bf16(a, b3, acc[3]);
    __syncthreads();                           // reads done before buf is reused
  }
#else
  // Fallback: cooperative staging with plain loads (no TDM builtin).
  for (int k = 0; k < Kdim; k += GEMM_KT) {
    for (int r = threadIdx.x; r < GEMM_BN; r += 128) {
      const u32x4* src = (const u32x4*)(W + (size_t)(nBlock + r) * Kdim + k);
      u32x4* dst = (u32x4*)&ldsB[0][r][0];
      dst[0] = src[0]; dst[1] = src[1]; dst[2] = src[2]; dst[3] = src[3];
    }
    __syncthreads();
    v16bf a  = load_a_frag(A + (size_t)mBase * Kdim + k, Kdim, lane);
    v16bf b0 = load_b_frag(&ldsB[0][wave * 64 +  0][0], GEMM_KT, lane);
    v16bf b1 = load_b_frag(&ldsB[0][wave * 64 + 16][0], GEMM_KT, lane);
    v16bf b2 = load_b_frag(&ldsB[0][wave * 64 + 32][0], GEMM_KT, lane);
    v16bf b3 = load_b_frag(&ldsB[0][wave * 64 + 48][0], GEMM_KT, lane);
    acc[0] = wmma_bf16(a, b0, acc[0]);
    acc[1] = wmma_bf16(a, b1, acc[1]);
    acc[2] = wmma_bf16(a, b2, acc[2]);
    acc[3] = wmma_bf16(a, b3, acc[3]);
    __syncthreads();
  }
#endif
}

// ---------------------------------------------------------------- convert
__global__ void __launch_bounds__(256)
convert_bf16_kernel(const float* __restrict__ in, unsigned short* __restrict__ out,
                    int nElem) {
  int i = blockIdx.x * blockDim.x + threadIdx.x;
  if (i < nElem) out[i] = f2bf(in[i]);
}

// ---------------------------------------------------------------- QKV GEMM
// qkv[r, e] = sum_d x[r, d] * w_qkv[e, d]   (r = s*B+b, e in [0, 3D))
// Scatter: Q,K -> [B,H,S,dk] bf16 ; V -> transposed [B,H,dk,S] bf16
__global__ void __launch_bounds__(128)
qkv_gemm_kernel(const unsigned short* __restrict__ xb,
                const unsigned short* __restrict__ wq,
                unsigned short* __restrict__ qout,
                unsigned short* __restrict__ kout,
                unsigned short* __restrict__ vtout) {
  __shared__ __align__(16) unsigned short ldsB[2][GEMM_BN][GEMM_KT];
  const int lane = threadIdx.x & 31;
  const int wave = threadIdx.x >> 5;
  const int half = lane >> 4;
  const int n    = lane & 15;
  const int mBase  = blockIdx.x * 16;
  const int nBlock = blockIdx.y * GEMM_BN;
  const int nBase  = nBlock + wave * 64;

  v8f acc[4] = {};
  gemm_mainloop(xb, wq, DMODEL, E3, mBase, nBlock, ldsB, lane, wave, acc);

  #pragma unroll
  for (int v = 0; v < 8; ++v) {
    const int r = mBase + v + half * 8;     // row = s*B + b
    const int s = r / BATCH;
    const int b = r % BATCH;
    #pragma unroll
    for (int j = 0; j < 4; ++j) {
      const int e   = nBase + j * 16 + n;
      const int i3  = e / DMODEL;           // 0=q 1=k 2=v
      const int rem = e % DMODEL;
      const int hh  = rem / DKH;
      const int dk  = rem % DKH;
      const unsigned short bfv = f2bf(acc[j][v]);
      const size_t bh = (size_t)b * NHEAD + hh;
      if (i3 == 0)      qout [(bh * S_LEN + s) * DKH + dk] = bfv;
      else if (i3 == 1) kout [(bh * S_LEN + s) * DKH + dk] = bfv;
      else              vtout[(bh * DKH  + dk) * S_LEN + s] = bfv;
    }
  }
}

// ---------------------------------------------------------------- attention
// Flash-attention: 8 waves/block; each wave owns 16 query rows; kv strips of 32.
__global__ void __launch_bounds__(256)
attention_kernel(const unsigned short* __restrict__ qb,
                 const unsigned short* __restrict__ kb,
                 const unsigned short* __restrict__ vt,
                 unsigned short* __restrict__ ob) {
  __shared__ unsigned short pTile[8][16][32];   // per-wave P staging (D->A relayout)

  const int lane = threadIdx.x & 31;
  const int wave = threadIdx.x >> 5;
  const int half = lane >> 4;
  const int n    = lane & 15;
  const int bh   = blockIdx.y;
  const int b    = bh / NHEAD;
  const int h    = bh % NHEAD;
  const int qBase = blockIdx.x * 128 + wave * 16;

  const unsigned short* qhead = qb + (size_t)bh * S_LEN * DKH;
  const unsigned short* khead = kb + (size_t)bh * S_LEN * DKH;
  const unsigned short* vhead = vt + (size_t)bh * DKH * S_LEN;

  // Q fragments for dk 0..31 and 32..63 (loaded once per wave)
  const v16bf qf0 = load_a_frag(qhead + (size_t)qBase * DKH +  0, DKH, lane);
  const v16bf qf1 = load_a_frag(qhead + (size_t)qBase * DKH + 32, DKH, lane);

  v8f acc0 = {}, acc1 = {}, acc2 = {}, acc3 = {};
  float rowMax[8], rowSum[8];
  #pragma unroll
  for (int v = 0; v < 8; ++v) { rowMax[v] = -__builtin_inff(); rowSum[v] = 0.f; }

  const float scale = 0.125f;   // 1/sqrt(64)

  for (int kvt = 0; kvt < qBase + 16; kvt += 32) {
    // prefetch next strip of K rows (contiguous: 32 rows x 128B)
    if (kvt + 32 < qBase + 16) {
      __builtin_prefetch(khead + (size_t)(kvt + 32) * DKH + lane * 64, 0, 1);
      __builtin_prefetch(vhead + (size_t)lane * S_LEN + (kvt + 32), 0, 1);
    }

    // ---- scores: S = Q * K^T  (two 16x16 tiles, K-dim = 64 split as 32+32)
    v8f s0 = {}, s1 = {};
    {
      v16bf b00 = load_b_frag(khead + (size_t)(kvt     ) * DKH +  0, DKH, lane);
      v16bf b01 = load_b_frag(khead + (size_t)(kvt     ) * DKH + 32, DKH, lane);
      v16bf b10 = load_b_frag(khead + (size_t)(kvt + 16) * DKH +  0, DKH, lane);
      v16bf b11 = load_b_frag(khead + (size_t)(kvt + 16) * DKH + 32, DKH, lane);
      s0 = wmma_bf16(qf0, b00, s0);
      s0 = wmma_bf16(qf1, b01, s0);
      s1 = wmma_bf16(qf0, b10, s1);
      s1 = wmma_bf16(qf1, b11, s1);
    }

    // ---- causal mask + online softmax (stats per q-row; halves independent)
    float rescale[8];
    #pragma unroll
    for (int v = 0; v < 8; ++v) {
      const int qrow = qBase + v + half * 8;
      const int c0 = kvt + n;
      const int c1 = kvt + 16 + n;
      float x0 = (c0 <= qrow) ? s0[v] * scale : -__builtin_inff();
      float x1 = (c1 <= qrow) ? s1[v] * scale : -__builtin_inff();
      float t = fmaxf(x0, x1);
      #pragma unroll
      for (int off = 1; off < 16; off <<= 1)
        t = fmaxf(t, __shfl_xor(t, off, 32));
      const float nm = fmaxf(rowMax[v], t);
      const float sc = __expf(rowMax[v] - nm);
      const float p0 = __expf(x0 - nm);
      const float p1 = __expf(x1 - nm);
      float ps = p0 + p1;
      #pragma unroll
      for (int off = 1; off < 16; off <<= 1)
        ps += __shfl_xor(ps, off, 32);
      rowSum[v] = rowSum[v] * sc + ps;
      rowMax[v] = nm;
      rescale[v] = sc;
      pTile[wave][v + half * 8][n]      = f2bf(p0);
      pTile[wave][v + half * 8][16 + n] = f2bf(p1);
    }
    #pragma unroll
    for (int v = 0; v < 8; ++v) {
      acc0[v] *= rescale[v];
      acc1[v] *= rescale[v];
      acc2[v] *= rescale[v];
      acc3[v] *= rescale[v];
    }

    // same-wave LDS RAW: wait for the P stores before re-loading as A-fragment
    asm volatile("s_wait_dscnt 0" ::: "memory");

    // ---- O += P * V   (A = P[16x32] from LDS, B from V^T rows = contiguous kv)
    v16bf pf  = load_a_frag(&pTile[wave][0][0], 32, lane);
    v16bf bv0 = load_b_frag(vhead + (size_t)( 0) * S_LEN + kvt, S_LEN, lane);
    v16bf bv1 = load_b_frag(vhead + (size_t)(16) * S_LEN + kvt, S_LEN, lane);
    v16bf bv2 = load_b_frag(vhead + (size_t)(32) * S_LEN + kvt, S_LEN, lane);
    v16bf bv3 = load_b_frag(vhead + (size_t)(48) * S_LEN + kvt, S_LEN, lane);
    acc0 = wmma_bf16(pf, bv0, acc0);
    acc1 = wmma_bf16(pf, bv1, acc1);
    acc2 = wmma_bf16(pf, bv2, acc2);
    acc3 = wmma_bf16(pf, bv3, acc3);
  }

  // ---- normalize and store O directly in [S,B,D] bf16 layout
  #pragma unroll
  for (int v = 0; v < 8; ++v) {
    const int qrow = qBase + v + half * 8;
    const float inv = 1.0f / rowSum[v];
    unsigned short* orow = ob + ((size_t)qrow * BATCH + b) * DMODEL + h * DKH;
    orow[ 0 + n] = f2bf(acc0[v] * inv);
    orow[16 + n] = f2bf(acc1[v] * inv);
    orow[32 + n] = f2bf(acc2[v] * inv);
    orow[48 + n] = f2bf(acc3[v] * inv);
  }
}

// ---------------------------------------------------------------- out GEMM
// out[r, e] = sum_d O[r, d] * w_out[e, d] + b_out[e]   (f32 output)
__global__ void __launch_bounds__(128)
out_gemm_kernel(const unsigned short* __restrict__ ab,
                const unsigned short* __restrict__ wo,
                const float* __restrict__ bias,
                float* __restrict__ out) {
  __shared__ __align__(16) unsigned short ldsB[2][GEMM_BN][GEMM_KT];
  const int lane = threadIdx.x & 31;
  const int wave = threadIdx.x >> 5;
  const int half = lane >> 4;
  const int n    = lane & 15;
  const int mBase  = blockIdx.x * 16;
  const int nBlock = blockIdx.y * GEMM_BN;
  const int nBase  = nBlock + wave * 64;

  v8f acc[4] = {};
  gemm_mainloop(ab, wo, DMODEL, DMODEL, mBase, nBlock, ldsB, lane, wave, acc);

  #pragma unroll
  for (int v = 0; v < 8; ++v) {
    const int r = mBase + v + half * 8;
    #pragma unroll
    for (int j = 0; j < 4; ++j) {
      const int col = nBase + j * 16 + n;
      out[(size_t)r * DMODEL + col] = acc[j][v] + bias[col];
    }
  }
}

// ---------------------------------------------------------------- launch
extern "C" void kernel_launch(void* const* d_in, const int* in_sizes, int n_in,
                              void* d_out, int out_size, void* d_ws, size_t ws_size,
                              hipStream_t stream) {
  const float* x     = (const float*)d_in[0];
  const float* w_qkv = (const float*)d_in[1];
  const float* w_out = (const float*)d_in[2];
  const float* b_out = (const float*)d_in[3];
  float* out = (float*)d_out;

  constexpr size_t szX    = (size_t)ROWS * DMODEL;               // 4M elems
  constexpr size_t szWq   = (size_t)E3 * DMODEL;                 // 3M elems
  constexpr size_t szWo   = (size_t)DMODEL * DMODEL;             // 1M elems
  constexpr size_t szHead = (size_t)BATCH * NHEAD * S_LEN * DKH; // 4M elems

  char* ws = (char*)d_ws;
  unsigned short* xb  = (unsigned short*)ws;                 ws += szX    * 2;
  unsigned short* wqb = (unsigned short*)ws;                 ws += szWq   * 2;
  unsigned short* wob = (unsigned short*)ws;                 ws += szWo   * 2;
  unsigned short* qb  = (unsigned short*)ws;                 ws += szHead * 2;
  unsigned short* kbp = (unsigned short*)ws;                 ws += szHead * 2;
  unsigned short* vtb = (unsigned short*)ws;                 ws += szHead * 2;
  unsigned short* ob  = (unsigned short*)ws;                 ws += szX    * 2;
  (void)ws_size; (void)in_sizes; (void)n_in; (void)out_size;

  convert_bf16_kernel<<<(unsigned)((szX  + 255) / 256), 256, 0, stream>>>(x,     xb,  (int)szX);
  convert_bf16_kernel<<<(unsigned)((szWq + 255) / 256), 256, 0, stream>>>(w_qkv, wqb, (int)szWq);
  convert_bf16_kernel<<<(unsigned)((szWo + 255) / 256), 256, 0, stream>>>(w_out, wob, (int)szWo);

  qkv_gemm_kernel<<<dim3(ROWS / 16, E3 / GEMM_BN), 128, 0, stream>>>(xb, wqb, qb, kbp, vtb);

  attention_kernel<<<dim3(S_LEN / 128, BATCH * NHEAD), 256, 0, stream>>>(qb, kbp, vtb, ob);

  out_gemm_kernel<<<dim3(ROWS / 16, DMODEL / GEMM_BN), 128, 0, stream>>>(ob, wob, b_out, out);
}
